// FlexibleMamba_77884936946211
// MI455X (gfx1250) — matrix operand order repaired and checked by reading, not compiled, not run on hardware
//
#include <hip/hip_runtime.h>

typedef __attribute__((ext_vector_type(16))) _Float16 v16h;
typedef __attribute__((ext_vector_type(8)))  _Float16 v8h;
typedef __attribute__((ext_vector_type(8)))  float    v8f;

#define DMODEL   256
#define DSTATE   64
#define DCONV    4
#define HEADDIM  128
#define DINNER   1024
#define NHEADS   8
#define CONVDIM  1152            // DINNER + 2*DSTATE
#define BATCH    8
#define SEQ      2048
#define ROWS     (BATCH*SEQ)     // 16384
#define DPROJ    2184            // 2*DINNER + 2*DSTATE + NHEADS
#define DPROJ_PAD 2192           // padded to multiple of 16
#define NCHUNK   16              // scan chunks per sequence
#define CLEN     (SEQ/NCHUNK)    // 128 steps per chunk
#define STATE_ELEMS (HEADDIM*DSTATE)   // 8192 floats per (b,h) state

union FragU { v16h v; v8h h[2]; };

// A-matrix 16x32 f16 fragment (ISA 7.12.2): lane m = lane%16,
// elems 0..7 = K[g*8 .. g*8+7], elems 8..15 = K[g*8+16 .. g*8+23], g = lane/16
__device__ __forceinline__ v16h load_fragA(const _Float16* __restrict__ p, int ld,
                                           int m, int kb, int g) {
  const _Float16* q = p + (size_t)m * ld + kb + g * 8;
  FragU u;
  u.h[0] = *(const v8h*)q;
  u.h[1] = *(const v8h*)(q + 16);
  return u.v;
}

// B-matrix 32x16 f16 fragment: lane n = lane%16,
// elems 0..7 = K[g*8 .. g*8+7], elems 8..15 = K[g*8+16 .. g*8+23], g = lane/16
__device__ __forceinline__ v16h load_fragB(const _Float16* __restrict__ p, int ld,
                                           int n, int kb, int g) {
  const _Float16* q = p + (size_t)n * ld + kb + g * 8;
  FragU u;
  u.h[0] = *(const v8h*)q;
  u.h[1] = *(const v8h*)(q + 16);
  return u.v;
}

#define WMMA(A, B, C) __builtin_amdgcn_wmma_f32_16x16x32_f16(false, (A), false, (B), (short)0, (C), false, false)

// C[M x N] = A[M x K] * B[N x K]^T  with 2-term f16 split precision.
// One wave (32 threads) per block computes a 32(M) x 16(N) tile.
__global__ void mamba_gemm_wmma(const _Float16* __restrict__ Ah, const _Float16* __restrict__ Al,
                                const _Float16* __restrict__ Bh, const _Float16* __restrict__ Bl,
                                float* __restrict__ C, int K, int ldc) {
  const int lane = threadIdx.x;
  const int lo = lane & 15;
  const int g  = lane >> 4;
  const int m0 = blockIdx.x * 32;
  const int n0 = blockIdx.y * 16;

  v8f acc0 = {};
  v8f acc1 = {};
  for (int kb = 0; kb < K; kb += 32) {
    v16h bh  = load_fragB(Bh, K, n0 + lo, kb, g);
    v16h bl  = load_fragB(Bl, K, n0 + lo, kb, g);
    v16h a0h = load_fragA(Ah, K, m0 + lo,      kb, g);
    v16h a0l = load_fragA(Al, K, m0 + lo,      kb, g);
    v16h a1h = load_fragA(Ah, K, m0 + 16 + lo, kb, g);
    v16h a1l = load_fragA(Al, K, m0 + 16 + lo, kb, g);
    acc0 = WMMA(a0h, bh, acc0);
    acc0 = WMMA(a0h, bl, acc0);
    acc0 = WMMA(a0l, bh, acc0);
    acc1 = WMMA(a1h, bh, acc1);
    acc1 = WMMA(a1h, bl, acc1);
    acc1 = WMMA(a1l, bh, acc1);
  }
  // C/D layout: elem r -> row (r + g*8), col = lane%16
#pragma unroll
  for (int r = 0; r < 8; ++r) {
    int row = m0 + r + g * 8;
    int col = n0 + lo;
    C[(size_t)row * ldc + col]        = acc0[r];
    C[(size_t)(row + 16) * ldc + col] = acc1[r];
  }
}

// fp32 -> (f16 hi, f16 lo) split
__global__ void split_f16_k(const float* __restrict__ src, _Float16* __restrict__ hi,
                            _Float16* __restrict__ lo, int n) {
  int i = blockIdx.x * 256 + threadIdx.x;
  if (i < n) {
    float v = src[i];
    _Float16 h = (_Float16)v;
    hi[i] = h;
    lo[i] = (_Float16)(v - (float)h);
  }
}

// same, but row-pad (zero rows >= rows_valid); cols unchanged
__global__ void split_f16_pad_k(const float* __restrict__ src, _Float16* __restrict__ hi,
                                _Float16* __restrict__ lo, int rows_valid, int cols, int ntot) {
  int i = blockIdx.x * 256 + threadIdx.x;
  if (i < ntot) {
    int r = i / cols;
    float v = (r < rows_valid) ? src[i] : 0.f;
    _Float16 h = (_Float16)v;
    hi[i] = h;
    lo[i] = (_Float16)(v - (float)h);
  }
}

// depthwise causal conv (K=4) + SiLU over the xBC slice of zxbcdt
__global__ void conv_silu_k(const float* __restrict__ zx, const float* __restrict__ cw,
                            const float* __restrict__ cb, float* __restrict__ xbc) {
  int idx = blockIdx.x * 256 + threadIdx.x;
  if (idx >= ROWS * CONVDIM) return;
  int c = idx % CONVDIM;
  int r = idx / CONVDIM;
  int b = r / SEQ, t = r % SEQ;
  float acc = cb[c];
#pragma unroll
  for (int k = 0; k < DCONV; ++k) {
    int tt = t + k - (DCONV - 1);
    if (tt >= 0)
      acc = fmaf(zx[(size_t)(b * SEQ + tt) * DPROJ_PAD + DINNER + c], cw[k * CONVDIM + c], acc);
  }
  xbc[idx] = acc / (1.f + __expf(-acc));     // SiLU
}

__global__ void dt_softplus_k(const float* __restrict__ zx, const float* __restrict__ dt_bias,
                              float* __restrict__ dts) {
  int idx = blockIdx.x * 256 + threadIdx.x;
  if (idx >= ROWS * NHEADS) return;
  int r = idx / NHEADS, h = idx % NHEADS;
  float v = zx[(size_t)r * DPROJ_PAD + DINNER + CONVDIM + h] + dt_bias[h];
  dts[idx] = (v > 20.f) ? v : log1pf(__expf(v));
}

// ---------------- chunk-parallel selective scan ----------------
// Recurrence h_t = dA_t * h_{t-1} + dtx_t (x) B_t is a linear scan; we chunk it.
// Pass 1: per (b,h,chunk) compute local end-state (h0=0) and decay product P.
// Pass 2: per (b,h) sequentially combine 16 chunk states (in place: slot <- init state).
// Pass 3: per (b,h,chunk) re-run chunk from its true init state, emit y.
// Thread mapping (all passes): p = tid&127, n in [ (tid>>7)*32, +32 );
// state storage index = j*256 + tid (coalesced).

__global__ void scan_pass1_k(const float* __restrict__ xbc, const float* __restrict__ dts,
                             const float* __restrict__ A_log,
                             float* __restrict__ states, float* __restrict__ dacum) {
  const int bh = blockIdx.x / NCHUNK;
  const int c  = blockIdx.x % NCHUNK;
  const int b  = bh / NHEADS;
  const int h  = bh % NHEADS;
  const int tid = threadIdx.x;
  const int p = tid & 127;
  const int nbase = (tid >> 7) * 32;

  __shared__ float sB[DSTATE];

  const float negA = -__expf(A_log[h]);
  float s[32];
#pragma unroll
  for (int j = 0; j < 32; ++j) s[j] = 0.f;
  float dAc = 1.f;

  const int t0 = c * CLEN;
  for (int t = t0; t < t0 + CLEN; ++t) {
    const int r = b * SEQ + t;
    const float* row = xbc + (size_t)r * CONVDIM;
    if (tid < 64) sB[tid] = row[DINNER + tid];
    const float dt_t = dts[r * NHEADS + h];
    const float dA   = __expf(dt_t * negA);
    const float dtx  = dt_t * row[h * HEADDIM + p];
    __builtin_prefetch(row + CONVDIM + DINNER + (tid & 63), 0, 1);  // next step's B
    __syncthreads();
#pragma unroll
    for (int j = 0; j < 32; ++j)
      s[j] = fmaf(s[j], dA, dtx * sB[nbase + j]);
    dAc *= dA;
    __syncthreads();
  }
  float* st = states + (size_t)blockIdx.x * STATE_ELEMS;
#pragma unroll
  for (int j = 0; j < 32; ++j) st[j * 256 + tid] = s[j];
  if (tid == 0) dacum[blockIdx.x] = dAc;
}

__global__ void scan_pass2_k(float* __restrict__ states, const float* __restrict__ dacum) {
  const int bh = blockIdx.x;        // 64 blocks
  const int tid = threadIdx.x;
  float hcur[32];
#pragma unroll
  for (int j = 0; j < 32; ++j) hcur[j] = 0.f;
  for (int c = 0; c < NCHUNK; ++c) {
    float* st = states + (size_t)(bh * NCHUNK + c) * STATE_ELEMS;
    const float P = dacum[bh * NCHUNK + c];
#pragma unroll
    for (int j = 0; j < 32; ++j) {
      const float L = st[j * 256 + tid];   // local end-state of chunk c
      st[j * 256 + tid] = hcur[j];         // becomes init state for chunk c
      hcur[j] = fmaf(P, hcur[j], L);
    }
  }
}

__global__ void scan_pass3_k(const float* __restrict__ xbc, const float* __restrict__ dts,
                             const float* __restrict__ A_log, const float* __restrict__ Dvec,
                             const float* __restrict__ states, float* __restrict__ y) {
  const int bh = blockIdx.x / NCHUNK;
  const int c  = blockIdx.x % NCHUNK;
  const int b  = bh / NHEADS;
  const int h  = bh % NHEADS;
  const int tid = threadIdx.x;
  const int p = tid & 127;
  const int nbase = (tid >> 7) * 32;

  __shared__ float sB[DSTATE];
  __shared__ float sC[DSTATE];
  __shared__ float red[256];

  const float negA = -__expf(A_log[h]);
  const float Dh = Dvec[h];
  const float* st = states + (size_t)blockIdx.x * STATE_ELEMS;
  float s[32];
#pragma unroll
  for (int j = 0; j < 32; ++j) s[j] = st[j * 256 + tid];

  const int t0 = c * CLEN;
  for (int t = t0; t < t0 + CLEN; ++t) {
    const int r = b * SEQ + t;
    const float* row = xbc + (size_t)r * CONVDIM;
    if (tid < 64)       sB[tid]      = row[DINNER + tid];
    else if (tid < 128) sC[tid - 64] = row[DINNER + DSTATE + (tid - 64)];
    const float dt_t = dts[r * NHEADS + h];
    const float dA   = __expf(dt_t * negA);
    const float xs   = row[h * HEADDIM + p];
    const float dtx  = dt_t * xs;
    __builtin_prefetch(row + CONVDIM + h * HEADDIM + p, 0, 1);   // next step's x slice
    __syncthreads();
    float acc = 0.f;
#pragma unroll
    for (int j = 0; j < 32; ++j) {
      const int n = nbase + j;
      s[j] = fmaf(s[j], dA, dtx * sB[n]);
      acc  = fmaf(s[j], sC[n], acc);
    }
    red[tid] = acc;
    __syncthreads();
    if (tid < 128)
      y[(size_t)r * DINNER + h * HEADDIM + p] = red[tid] + red[tid + 128] + Dh * xs;
  }
}

// gate with SiLU(z), RMS-norm over DINNER, emit f16 hi/lo split for out-proj GEMM
__global__ void gate_rmsnorm_k(const float* __restrict__ y, const float* __restrict__ zx,
                               const float* __restrict__ norm_w,
                               _Float16* __restrict__ gh, _Float16* __restrict__ gl) {
  const int r = blockIdx.x;
  const int tid = threadIdx.x;
  const float* yr = y + (size_t)r * DINNER;
  const float* zr = zx + (size_t)r * DPROJ_PAD;
  float v[4];
  float ss = 0.f;
#pragma unroll
  for (int j = 0; j < 4; ++j) {
    const int c = tid + j * 256;
    const float z = zr[c];
    const float gsig = z / (1.f + __expf(-z));
    const float t = yr[c] * gsig;
    v[j] = t;
    ss += t * t;
  }
  __shared__ float rb[256];
  rb[tid] = ss;
  __syncthreads();
  for (int sgt = 128; sgt > 0; sgt >>= 1) {
    if (tid < sgt) rb[tid] += rb[tid + sgt];
    __syncthreads();
  }
  const float scale = rsqrtf(rb[0] / (float)DINNER + 1e-5f);
#pragma unroll
  for (int j = 0; j < 4; ++j) {
    const int c = tid + j * 256;
    const float o = v[j] * scale * norm_w[c];
    const _Float16 hh = (_Float16)o;
    gh[(size_t)r * DINNER + c] = hh;
    gl[(size_t)r * DINNER + c] = (_Float16)(o - (float)hh);
  }
}

extern "C" void kernel_launch(void* const* d_in, const int* in_sizes, int n_in,
                              void* d_out, int out_size, void* d_ws, size_t ws_size,
                              hipStream_t stream) {
  const float* x       = (const float*)d_in[0];
  const float* w_in    = (const float*)d_in[1];
  const float* conv_w  = (const float*)d_in[2];
  const float* conv_b  = (const float*)d_in[3];
  const float* dt_bias = (const float*)d_in[4];
  const float* A_log   = (const float*)d_in[5];
  const float* Dv      = (const float*)d_in[6];
  const float* norm_w  = (const float*)d_in[7];
  const float* w_out   = (const float*)d_in[8];
  float* out = (float*)d_out;

  char* ws = (char*)d_ws;
  size_t off = 0;
  auto take = [&](size_t bytes) -> char* {
    char* p = ws + off;
    off = (off + bytes + 255) & ~(size_t)255;
    return p;
  };

  _Float16* xh     = (_Float16*)take((size_t)ROWS * DMODEL * 2);
  _Float16* xl     = (_Float16*)take((size_t)ROWS * DMODEL * 2);
  _Float16* wih    = (_Float16*)take((size_t)DPROJ_PAD * DMODEL * 2);
  _Float16* wil    = (_Float16*)take((size_t)DPROJ_PAD * DMODEL * 2);
  _Float16* woh    = (_Float16*)take((size_t)DMODEL * DINNER * 2);
  _Float16* wol    = (_Float16*)take((size_t)DMODEL * DINNER * 2);
  float*    zx     = (float*)take((size_t)ROWS * DPROJ_PAD * 4);
  float*    xbc    = (float*)take((size_t)ROWS * CONVDIM * 4);
  float*    dts    = (float*)take((size_t)ROWS * NHEADS * 4);
  float*    yf     = (float*)take((size_t)ROWS * DINNER * 4);
  _Float16* ygh    = (_Float16*)take((size_t)ROWS * DINNER * 2);
  _Float16* ygl    = (_Float16*)take((size_t)ROWS * DINNER * 2);
  float*    states = (float*)take((size_t)BATCH * NHEADS * NCHUNK * STATE_ELEMS * 4);
  float*    dacum  = (float*)take((size_t)BATCH * NHEADS * NCHUNK * 4);
  (void)ws_size; (void)in_sizes; (void)n_in; (void)out_size;

  // 0) split fp32 inputs to f16 hi/lo
  {
    int n = ROWS * DMODEL;
    split_f16_k<<<(n + 255) / 256, 256, 0, stream>>>(x, xh, xl, n);
  }
  {
    int n = DPROJ_PAD * DMODEL;
    split_f16_pad_k<<<(n + 255) / 256, 256, 0, stream>>>(w_in, wih, wil, DPROJ, DMODEL, n);
  }
  {
    int n = DMODEL * DINNER;
    split_f16_k<<<(n + 255) / 256, 256, 0, stream>>>(w_out, woh, wol, n);
  }

  // 1) in-projection GEMM: zx[ROWS x DPROJ_PAD] = x @ w_in^T   (WMMA f16-split)
  {
    dim3 grid(ROWS / 32, DPROJ_PAD / 16);
    mamba_gemm_wmma<<<grid, 32, 0, stream>>>(xh, xl, wih, wil, zx, DMODEL, DPROJ_PAD);
  }

  // 2) causal depthwise conv + SiLU
  {
    int n = ROWS * CONVDIM;
    conv_silu_k<<<(n + 255) / 256, 256, 0, stream>>>(zx, conv_w, conv_b, xbc);
  }

  // 3) dt = softplus(dt + dt_bias)
  {
    int n = ROWS * NHEADS;
    dt_softplus_k<<<(n + 255) / 256, 256, 0, stream>>>(zx, dt_bias, dts);
  }

  // 4) chunk-parallel selective scan (3 passes)
  scan_pass1_k<<<BATCH * NHEADS * NCHUNK, 256, 0, stream>>>(xbc, dts, A_log, states, dacum);
  scan_pass2_k<<<BATCH * NHEADS, 256, 0, stream>>>(states, dacum);
  scan_pass3_k<<<BATCH * NHEADS * NCHUNK, 256, 0, stream>>>(xbc, dts, A_log, Dv, states, yf);

  // 5) gate + RMS-norm, emit f16 hi/lo
  gate_rmsnorm_k<<<ROWS, 256, 0, stream>>>(yf, zx, norm_w, ygh, ygl);

  // 6) out-projection GEMM: out[ROWS x DMODEL] = yg @ w_out^T
  {
    dim3 grid(ROWS / 32, DMODEL / 16);
    mamba_gemm_wmma<<<grid, 32, 0, stream>>>(ygh, ygl, woh, wol, out, DINNER, DMODEL);
  }
}
